// ProbAttention_69552700392014
// MI455X (gfx1250) — matrix operand order repaired and checked by reading, not compiled, not run on hardware
//
#include <hip/hip_runtime.h>

// ---------------------------------------------------------------------------
// ProbSparse attention (Informer) for MI455X / gfx1250, wave32 + WMMA f16.
// B=8, L=2048, H=8, D=64, sample_k = u = 40 (u read from in_sizes at launch).
// ---------------------------------------------------------------------------

typedef __attribute__((ext_vector_type(16))) _Float16 v16h;
typedef __attribute__((ext_vector_type(8)))  float    v8f;
typedef __attribute__((ext_vector_type(4)))  float    v4f;

#define BATCH 8
#define L_SEQ 2048
#define DHEAD 64
#define KT    32     // keys processed per flash iteration
#define UPAD  48     // 40 selected queries padded to 3 row-tiles of 16
#define TPB4  96     // 3 waves per block in the attention kernel

// ---- WMMA fragment loaders (contiguous k = 16h + s, not the ISA table order; A and B alike, wave32) ------
// A matrix 16x32 (MxK), row-major source [row][ld]:
//   lanes 0-15 : M=lane,   halves 0-15 -> K 0-15
//   lanes16-31 : M=lane-16,halves 0-15 -> K 16-31
__device__ inline v16h load_a_frag(const _Float16* src, int ld, int kbase) {
  const int lane = threadIdx.x & 31;
  const int row  = lane & 15;
  const bool hi  = lane >= 16;
  const _Float16* p = src + row * ld + kbase;
  const int o1 = hi ? 16 : 0;
  v16h a;
#pragma unroll
  for (int i = 0; i < 16; ++i) a[i] = p[o1 + i];
  return a;
}

// B matrix 32x16 (KxN), source stored N-major: [N][ldk] (i.e. row n holds the
// K-dim contiguously):
//   lanes 0-15 : N=lane,    halves 0-15 -> K 0-15
//   lanes16-31 : N=lane-16, halves 0-15 -> K 16-31
__device__ inline v16h load_b_frag(const _Float16* src, int ldk, int nbase, int kbase) {
  const int lane = threadIdx.x & 31;
  const _Float16* p = src + (nbase + (lane & 15)) * ldk + kbase + ((lane >= 16) ? 16 : 0);
  v16h b;
#pragma unroll
  for (int i = 0; i < 16; ++i) b[i] = p[i];
  return b;
}

__device__ inline float dot4(float4 a, float4 b) {
  return a.x * b.x + a.y * b.y + a.z * b.z + a.w * b.w;
}

// ---------------------------------------------------------------------------
// Kernel 1: sampled sparsity metric  M[b,h,q] = max_s(QK_s) - mean_s(QK_s)
// one thread per (b,h,q); S gathered key rows of 64 floats each.
// ---------------------------------------------------------------------------
__global__ void __launch_bounds__(256)
sampled_m_kernel(const float* __restrict__ Q, const float* __restrict__ K,
                 const int* __restrict__ idx, float* __restrict__ Mbuf,
                 int H, int S) {
  const int gid = blockIdx.x * blockDim.x + threadIdx.x;   // (b*H + h)*L + q
  const int q   = gid % L_SEQ;
  const int bh  = gid / L_SEQ;
  const int b   = bh / H, h = bh % H;

  const float4* qv = (const float4*)(Q + (((size_t)b * L_SEQ + q) * H + h) * DHEAD);
  float4 qr[16];
#pragma unroll
  for (int j = 0; j < 16; ++j) qr[j] = qv[j];

  float mx = -3.0e38f, sum = 0.f;
  for (int s = 0; s < S; ++s) {
    const int ki = idx[q * S + s];
    const float4* kv = (const float4*)(K + (((size_t)b * L_SEQ + ki) * H + h) * DHEAD);
    float d = 0.f;
#pragma unroll
    for (int j = 0; j < 16; ++j) d += dot4(qr[j], kv[j]);
    mx = fmaxf(mx, d);
    sum += d;
  }
  Mbuf[gid] = mx - sum / (float)S;
}

// ---------------------------------------------------------------------------
// Kernel 2: per-(b,h) top-u of M over L=2048 (iterative block argmax,
// tie-break on lowest index like jax.lax.top_k).
// ---------------------------------------------------------------------------
__global__ void __launch_bounds__(256)
topk_kernel(const float* __restrict__ Mbuf, int* __restrict__ topk, int u) {
  __shared__ float vals[L_SEQ];
  __shared__ float rv[256];
  __shared__ int   ri[256];
  const int bh = blockIdx.x;
  const int tid = threadIdx.x;

  for (int i = tid; i < L_SEQ; i += 256) vals[i] = Mbuf[(size_t)bh * L_SEQ + i];
  __syncthreads();

  for (int it = 0; it < u; ++it) {
    float best = -3.0e38f;
    int   bi   = 0x7FFFFFFF;
    for (int i = tid; i < L_SEQ; i += 256) {
      const float v = vals[i];
      if (v > best || (v == best && i < bi)) { best = v; bi = i; }
    }
    rv[tid] = best; ri[tid] = bi;
    __syncthreads();
    for (int s = 128; s > 0; s >>= 1) {
      if (tid < s) {
        if (rv[tid + s] > rv[tid] ||
            (rv[tid + s] == rv[tid] && ri[tid + s] < ri[tid])) {
          rv[tid] = rv[tid + s]; ri[tid] = ri[tid + s];
        }
      }
      __syncthreads();
    }
    if (tid == 0) { topk[bh * u + it] = ri[0]; vals[ri[0]] = -3.0e38f; }
    __syncthreads();
  }
}

// ---------------------------------------------------------------------------
// Kernel 3a: vmean[b,h,d] = mean over keys of V.  One block per (b,h),
// 256 threads = 4 key-partitions x 64 d, coalesced reads, LDS reduce.
// ---------------------------------------------------------------------------
__global__ void __launch_bounds__(256)
vmean_kernel(const float* __restrict__ V, float* __restrict__ vmean, int H) {
  __shared__ float part[4][DHEAD];
  const int bh = blockIdx.x;
  const int b = bh / H, h = bh % H;
  const int d  = threadIdx.x & 63;
  const int kp = threadIdx.x >> 6;   // 0..3
  float acc = 0.f;
  for (int k = kp; k < L_SEQ; k += 4)
    acc += V[(((size_t)b * L_SEQ + k) * H + h) * DHEAD + d];
  part[kp][d] = acc;
  __syncthreads();
  if (kp == 0) {
    const float m = (part[0][d] + part[1][d] + part[2][d] + part[3][d]) *
                    (1.0f / (float)L_SEQ);
    vmean[bh * DHEAD + d] = m;
  }
}

// ---------------------------------------------------------------------------
// Kernel 3b: broadcast vmean over the whole output (the non-selected queries
// keep this baseline).  Grid-strided 16B stores with NT hint: this is a
// 33.5 MB write-once stream -- keep it from displacing K/V in L2, which the
// attention kernel re-reads.
// ---------------------------------------------------------------------------
__global__ void __launch_bounds__(256)
fill_kernel(const float* __restrict__ vmean, float* __restrict__ out, int H) {
  const size_t total4 = (size_t)BATCH * L_SEQ * H * DHEAD / 4;
  const size_t stride = (size_t)gridDim.x * blockDim.x;
  for (size_t i = (size_t)blockIdx.x * blockDim.x + threadIdx.x; i < total4;
       i += stride) {
    const size_t e   = i * 4;
    const int    d   = (int)(e & (DHEAD - 1));      // d within head (mult of 4)
    const size_t bqh = e >> 6;                      // (b*L+q)*H + h
    const int    h   = (int)(bqh % H);
    const size_t b   = bqh / ((size_t)L_SEQ * H);
    const v4f m = *(const v4f*)(vmean + ((b * H + h) * DHEAD + d));
    __builtin_nontemporal_store(m, (v4f*)out + i);
  }
}

// ---------------------------------------------------------------------------
// Kernel 4: dense attention for the u selected queries, flash-style online
// softmax, WMMA f32_16x16x32_f16 for both GEMMs. One block per (b,h),
// 3 waves x 16 query rows.
// ---------------------------------------------------------------------------
__global__ void __launch_bounds__(TPB4)
prob_attn_topq_kernel(const float* __restrict__ Q, const float* __restrict__ K,
                      const float* __restrict__ V, const int* __restrict__ topk,
                      float* __restrict__ out, int H, int u) {
  __shared__ _Float16 sQ[UPAD * DHEAD];          // [row][d]        6 KB
  __shared__ _Float16 sK[KT * DHEAD];            // [key][d]        4 KB
  __shared__ _Float16 sVt[DHEAD * KT];           // [d][key]        4 KB
  __shared__ _Float16 sP[(TPB4 / 32) * 16 * KT]; // per-wave P      3 KB

  const int bh = blockIdx.x;
  const int b = bh / H, h = bh % H;
  const int tid  = threadIdx.x;
  const int wave = tid >> 5;
  const int lane = tid & 31;
  const int nrow = lane & 15;
  const bool hiw = lane >= 16;

  // ---- stage Q_reduce (f16), zero-pad rows >= u
  for (int e = tid; e < UPAD * DHEAD; e += TPB4) {
    const int ui = e / DHEAD, d = e % DHEAD;
    float val = 0.f;
    if (ui < u) {
      const int qi = topk[bh * u + ui];
      val = Q[(((size_t)b * L_SEQ + qi) * H + h) * DHEAD + d];
    }
    sQ[e] = (_Float16)val;
  }
  __syncthreads();

  const v16h aQ0 = load_a_frag(sQ + wave * 16 * DHEAD, DHEAD, 0);
  const v16h aQ1 = load_a_frag(sQ + wave * 16 * DHEAD, DHEAD, 32);

  v8f O0 = {}, O1 = {}, O2 = {}, O3 = {};
  float mrow[8], lrow[8];
#pragma unroll
  for (int r = 0; r < 8; ++r) { mrow[r] = -3.0e38f; lrow[r] = 0.f; }

  _Float16* myP = sP + wave * 16 * KT;

  for (int kt = 0; kt < L_SEQ; kt += KT) {
    // cooperative K tile load + V tile transposed load (f32 -> f16)
    for (int e = tid; e < KT * DHEAD; e += TPB4) {
      const int kk = e / DHEAD, d = e % DHEAD;
      const size_t base = (((size_t)b * L_SEQ + (kt + kk)) * H + h) * DHEAD + d;
      sK[e] = (_Float16)K[base];
      sVt[d * KT + kk] = (_Float16)V[base];
    }
    __syncthreads();

    // S = Q * K^T : 16 rows x 32 keys, f32 accumulate
    v8f C0 = {}, C1 = {};
    {
      v16h bk = load_b_frag(sK, DHEAD, 0, 0);
      C0 = __builtin_amdgcn_wmma_f32_16x16x32_f16(false, aQ0, false, bk, (short)0, C0, false, false);
      bk = load_b_frag(sK, DHEAD, 0, 32);
      C0 = __builtin_amdgcn_wmma_f32_16x16x32_f16(false, aQ1, false, bk, (short)0, C0, false, false);
      bk = load_b_frag(sK, DHEAD, 16, 0);
      C1 = __builtin_amdgcn_wmma_f32_16x16x32_f16(false, aQ0, false, bk, (short)0, C1, false, false);
      bk = load_b_frag(sK, DHEAD, 16, 32);
      C1 = __builtin_amdgcn_wmma_f32_16x16x32_f16(false, aQ1, false, bk, (short)0, C1, false, false);
    }

    // online softmax; scale = 1/sqrt(64) = 0.125
#pragma unroll
    for (int r = 0; r < 8; ++r) {
      const float s0 = C0[r] * 0.125f;
      const float s1 = C1[r] * 0.125f;
      float mx = fmaxf(s0, s1);
#pragma unroll
      for (int msk = 8; msk >= 1; msk >>= 1) mx = fmaxf(mx, __shfl_xor(mx, msk, 32));
      const float mnew  = fmaxf(mrow[r], mx);
      const float alpha = __expf(mrow[r] - mnew);
      const float p0 = __expf(s0 - mnew);
      const float p1 = __expf(s1 - mnew);
      float rs = p0 + p1;
#pragma unroll
      for (int msk = 8; msk >= 1; msk >>= 1) rs += __shfl_xor(rs, msk, 32);
      lrow[r] = lrow[r] * alpha + rs;
      mrow[r] = mnew;
      O0[r] *= alpha; O1[r] *= alpha; O2[r] *= alpha; O3[r] *= alpha;
      const int prow = r + (hiw ? 8 : 0);
      myP[prow * KT + nrow]      = (_Float16)p0;   // C-layout -> LDS
      myP[prow * KT + 16 + nrow] = (_Float16)p1;
    }

    // O += P * V   (A = P 16x32, B = V 32x64 via transposed LDS tile)
    {
      const v16h aP = load_a_frag(myP, KT, 0);
      v16h bv = load_b_frag(sVt, KT, 0, 0);
      O0 = __builtin_amdgcn_wmma_f32_16x16x32_f16(false, aP, false, bv, (short)0, O0, false, false);
      bv = load_b_frag(sVt, KT, 16, 0);
      O1 = __builtin_amdgcn_wmma_f32_16x16x32_f16(false, aP, false, bv, (short)0, O1, false, false);
      bv = load_b_frag(sVt, KT, 32, 0);
      O2 = __builtin_amdgcn_wmma_f32_16x16x32_f16(false, aP, false, bv, (short)0, O2, false, false);
      bv = load_b_frag(sVt, KT, 48, 0);
      O3 = __builtin_amdgcn_wmma_f32_16x16x32_f16(false, aP, false, bv, (short)0, O3, false, false);
    }
    __syncthreads();
  }

  // normalize and scatter into the broadcast-mean context
#pragma unroll
  for (int r = 0; r < 8; ++r) {
    const int row = r + (hiw ? 8 : 0);
    const int ui  = wave * 16 + row;
    if (ui < u) {
      const int qi  = topk[bh * u + ui];
      const float inv = 1.0f / lrow[r];
      const size_t base = (((size_t)b * L_SEQ + qi) * H + h) * DHEAD;
      out[base + 0 * 16 + nrow] = O0[r] * inv;
      out[base + 1 * 16 + nrow] = O1[r] * inv;
      out[base + 2 * 16 + nrow] = O2[r] * inv;
      out[base + 3 * 16 + nrow] = O3[r] * inv;
    }
  }
}

// ---------------------------------------------------------------------------
extern "C" void kernel_launch(void* const* d_in, const int* in_sizes, int n_in,
                              void* d_out, int out_size, void* d_ws, size_t ws_size,
                              hipStream_t stream) {
  const int B = BATCH, H = 8;
  const float* Q   = (const float*)d_in[0];
  const float* K   = (const float*)d_in[1];
  const float* V   = (const float*)d_in[2];
  const int*   idx = (const int*)d_in[3];
  float* out = (float*)d_out;

  const int S = in_sizes[3] / L_SEQ;  // sample_k (= 40)
  const int u = S;                    // u uses the same formula in the reference

  char* ws = (char*)d_ws;
  float* Mbuf   = (float*)ws;                       ws += (size_t)B * H * L_SEQ * sizeof(float);
  int*   topbuf = (int*)ws;                         ws += (size_t)B * H * u * sizeof(int);
  float* vmean  = (float*)ws;                       // B*H*DHEAD floats

  const int BHL = B * H * L_SEQ;
  sampled_m_kernel<<<BHL / 256, 256, 0, stream>>>(Q, K, idx, Mbuf, H, S);
  topk_kernel<<<B * H, 256, 0, stream>>>(Mbuf, topbuf, u);
  vmean_kernel<<<B * H, 256, 0, stream>>>(V, vmean, H);
  fill_kernel<<<4096, 256, 0, stream>>>(vmean, out, H);
  prob_attn_topq_kernel<<<B * H, TPB4, 0, stream>>>(Q, K, V, topbuf, out, H, u);
}
